// mainModel_7980049236046
// MI455X (gfx1250) — compile-verified, matrix operand order checked by reading, not run on hardware
//
#include <hip/hip_runtime.h>

typedef __bf16 bhalf;
typedef __attribute__((ext_vector_type(16))) __bf16 v16bf;
typedef __attribute__((ext_vector_type(8)))  __bf16 v8bf;
typedef __attribute__((ext_vector_type(8)))  float   v8f;

// ---------------------------------------------------------------------------
// WMMA helpers (CDNA5: V_WMMA_F32_16X16X32_BF16, wave32)
// ---------------------------------------------------------------------------
static __device__ __forceinline__ v8f wmma_bf16(v16bf a, v16bf b, v8f c) {
  // (neg_a, A, neg_b, B, c_mod, C, reuse_a, reuse_b)
  return __builtin_amdgcn_wmma_f32_16x16x32_bf16(false, a, false, b, (short)0, c,
                                                 false, false);
}

// Load a 16x32 bf16 fragment from row-major storage with leading dim `ld`.
// Serves A ([M x K] row-major) and B when B is stored [N x K] row-major
// (lane <-> output column, rows contiguous => 2x16B loads per lane).
// Layout per ISA 7.12.2: lanes 0-15 row r: K 0..7 & 16..23; lanes 16-31: 8..15 & 24..31.
static __device__ __forceinline__ v16bf load_frag(const bhalf* __restrict__ base, int ld) {
  const int lane = threadIdx.x & 31;
  const int half = lane >> 4;
  const int r    = lane & 15;
  const bhalf* p = base + (size_t)r * (size_t)ld + half * 8;
  v8bf lo = *(const v8bf*)(p);
  v8bf hi = *(const v8bf*)(p + 16);
  v16bf o;
#pragma unroll
  for (int i = 0; i < 8; ++i) { o[i] = lo[i]; o[i + 8] = hi[i]; }
  return o;
}

static __device__ __forceinline__ float apply_act(float x, int act) {
  if (act == 1) return fmaxf(x, 0.f);                       // relu
  if (act == 2) return x > 0.f ? x : (__expf(x) - 1.f);     // elu
  return x;
}

static __device__ __forceinline__ void store_tile(
    v8f acc, int m0, int n0, const float* __restrict__ bias,
    float* __restrict__ C, bhalf* __restrict__ Cbf, int ldcb, int N, int act) {
  const int lane = threadIdx.x & 31;
  const int half = lane >> 4, r = lane & 15;
  const int col = n0 + r;
  const float bv = bias ? bias[col] : 0.f;
#pragma unroll
  for (int v = 0; v < 8; ++v) {
    const int row = m0 + half * 8 + v;   // C/D layout: lanes16-31 -> rows +8
    float x = apply_act(acc[v] + bv, act);
    if (C)   C[(size_t)row * N + col] = x;
    if (Cbf) Cbf[(size_t)row * ldcb + col] = (bhalf)x;
  }
}

// ---------------------------------------------------------------------------
// Generic GEMM: C[M,N] = act(A[M,K] * B^T + bias), B stored [N,K] row-major.
// 256 threads = 8 waves; wave tile 32x32 (2x2 WMMA frags); block tile 64x128.
// M,K multiples of 32; N multiple of 32 (<=128 per grid.x step).
// ---------------------------------------------------------------------------
__global__ __launch_bounds__(256) void gemm_bf16(
    const bhalf* __restrict__ A, const bhalf* __restrict__ B,
    const float* __restrict__ bias,
    float* __restrict__ C, bhalf* __restrict__ Cbf, int ldcb,
    int M, int N, int K, int act)
{
  const int wave = threadIdx.x >> 5;
  const int wm = wave >> 2, wn = wave & 3;
  const int m0 = blockIdx.y * 64 + wm * 32;
  const int n0 = blockIdx.x * 128 + wn * 32;
  if (m0 >= M || n0 >= N) return;

  const bhalf* Ap0 = A + (size_t)m0 * K;
  const bhalf* Ap1 = A + (size_t)(m0 + 16) * K;
  const bhalf* Bp0 = B + (size_t)n0 * K;
  const bhalf* Bp1 = B + (size_t)(n0 + 16) * K;

  // Pull the weight panel (rows reused by every block) through GL2 early, and
  // warm the next block-row of A (streamed once). Emits global_prefetch_b8.
  {
    const int lane = threadIdx.x & 31;
    __builtin_prefetch(Bp0 + (size_t)lane * K, 0, 1);
    __builtin_prefetch(Bp1 + (size_t)lane * K, 0, 1);
    __builtin_prefetch(Ap0 + (size_t)(lane + 32) * K, 0, 0);
  }

  v8f acc00 = {}, acc01 = {}, acc10 = {}, acc11 = {};
  for (int k0 = 0; k0 < K; k0 += 32) {
    v16bf a0 = load_frag(Ap0 + k0, K);
    v16bf a1 = load_frag(Ap1 + k0, K);
    v16bf b0 = load_frag(Bp0 + k0, K);
    v16bf b1 = load_frag(Bp1 + k0, K);
    acc00 = wmma_bf16(a0, b0, acc00);
    acc01 = wmma_bf16(a0, b1, acc01);
    acc10 = wmma_bf16(a1, b0, acc10);
    acc11 = wmma_bf16(a1, b1, acc11);
  }
  store_tile(acc00, m0,      n0,      bias, C, Cbf, ldcb, N, act);
  store_tile(acc01, m0,      n0 + 16, bias, C, Cbf, ldcb, N, act);
  store_tile(acc10, m0 + 16, n0,      bias, C, Cbf, ldcb, N, act);
  store_tile(acc11, m0 + 16, n0 + 16, bias, C, Cbf, ldcb, N, act);
}

// ---------------------------------------------------------------------------
// GAT-style attention for one (batch b, head h): L=64, hd=128.
// sl/sr via leaky-dot; softmax; ctx = score(64x64) @ fr(64x128) via WMMA/LDS.
// block = 64 threads (2 waves); grid.x = B * NHEADS.
// ---------------------------------------------------------------------------
__global__ __launch_bounds__(64) void attn_kernel(
    const float* __restrict__ fl, const float* __restrict__ fr,
    const float* __restrict__ al, const float* __restrict__ ar,
    bhalf* __restrict__ hsa_bf)
{
  __shared__ __align__(16) bhalf s_sc[64 * 64];    // softmax probs, row-major
  __shared__ __align__(16) bhalf s_frT[128 * 64];  // fr^T: [n][k]
  __shared__ float s_al[128], s_ar[128], s_sr[64];

  const int bh = blockIdx.x;
  const int b = bh >> 1, hsel = bh & 1;
  const int t = threadIdx.x;                       // 0..63 (row id)

  s_al[t] = al[t];  s_al[t + 64] = al[t + 64];
  s_ar[t] = ar[t];  s_ar[t + 64] = ar[t + 64];
  __syncthreads();

  const size_t rowbase = ((size_t)(b * 64 + t)) * 256 + (size_t)hsel * 128;
  const float* flr = fl + rowbase;
  const float* frr = fr + rowbase;
  float accl = 0.f, accr = 0.f;
#pragma unroll 8
  for (int c = 0; c < 128; ++c) {
    float xl = flr[c]; xl = xl >= 0.f ? xl : 0.01f * xl;
    float xr = frr[c]; float xrl = xr >= 0.f ? xr : 0.01f * xr;
    accl += xl * s_al[c];
    accr += xrl * s_ar[c];
    s_frT[c * 64 + t] = (bhalf)xr;                 // raw fr, transposed
  }
  s_sr[t] = accr;
  __syncthreads();

  // softmax_j(sl[t] + sr[j])
  float mx = -3.0e38f;
#pragma unroll
  for (int j = 0; j < 64; ++j) mx = fmaxf(mx, accl + s_sr[j]);
  float pr[64]; float sum = 0.f;
#pragma unroll
  for (int j = 0; j < 64; ++j) { float p = __expf(accl + s_sr[j] - mx); pr[j] = p; sum += p; }
  const float inv = 1.f / sum;
#pragma unroll
  for (int j = 0; j < 64; ++j) s_sc[t * 64 + j] = (bhalf)(pr[j] * inv);
  __syncthreads();

  // ctx = score @ frSeg : wave w covers output cols [w*64, w*64+64)
  const int wave = t >> 5, lane = t & 31;
  const int half = lane >> 4, r = lane & 15;
#pragma unroll
  for (int tm = 0; tm < 4; ++tm) {
#pragma unroll
    for (int tn = 0; tn < 4; ++tn) {
      const int m0 = tm * 16;
      const int n0 = wave * 64 + tn * 16;
      v8f acc = {};
#pragma unroll
      for (int k0 = 0; k0 < 64; k0 += 32) {
        v16bf a  = load_frag(&s_sc[m0 * 64 + k0], 64);
        v16bf bb = load_frag(&s_frT[n0 * 64 + k0], 64);
        acc = wmma_bf16(a, bb, acc);
      }
      const int col = n0 + r;
#pragma unroll
      for (int v = 0; v < 8; ++v) {
        const int row = m0 + half * 8 + v;
        hsa_bf[((size_t)(b * 64 + row)) * 256 + (size_t)hsel * 128 + col] = (bhalf)acc[v];
      }
    }
  }
}

// ---------------------------------------------------------------------------
// LayerNorm(residual): h = LN(h + t) * g + b; wave-per-row (wave32 shuffles).
// ---------------------------------------------------------------------------
__global__ __launch_bounds__(256) void ln_residual(
    float* __restrict__ h, const float* __restrict__ tres,
    const float* __restrict__ g, const float* __restrict__ bta,
    bhalf* __restrict__ h_bf, int R)
{
  const int wave = threadIdx.x >> 5, lane = threadIdx.x & 31;
  const int row = blockIdx.x * 8 + wave;
  if (row >= R) return;
  const size_t base = (size_t)row * 256;
  float x[8]; float s = 0.f;
#pragma unroll
  for (int i = 0; i < 8; ++i) {
    const int c = i * 32 + lane;
    x[i] = h[base + c] + tres[base + c];
    s += x[i];
  }
#pragma unroll
  for (int o = 16; o > 0; o >>= 1) s += __shfl_xor(s, o, 32);
  const float mean = s * (1.f / 256.f);
  float v = 0.f;
#pragma unroll
  for (int i = 0; i < 8; ++i) { float d = x[i] - mean; v += d * d; }
#pragma unroll
  for (int o = 16; o > 0; o >>= 1) v += __shfl_xor(v, o, 32);
  const float inv = rsqrtf(v * (1.f / 256.f) + 1e-5f);
#pragma unroll
  for (int i = 0; i < 8; ++i) {
    const int c = i * 32 + lane;
    const float y = (x[i] - mean) * inv * g[c] + bta[c];
    h[base + c] = y;
    h_bf[base + c] = (bhalf)y;
  }
}

// ---------------------------------------------------------------------------
// Final head: out[b] = sigmoid(dot(m2[b,0:128], w) + b3)
// ---------------------------------------------------------------------------
__global__ __launch_bounds__(256) void head_kernel(
    const float* __restrict__ m2, const float* __restrict__ w,
    const float* __restrict__ b3, float* __restrict__ out, int Bn)
{
  const int wave = threadIdx.x >> 5, lane = threadIdx.x & 31;
  const int row = blockIdx.x * 8 + wave;
  if (row >= Bn) return;
  float s = 0.f;
#pragma unroll
  for (int i = 0; i < 4; ++i) { int c = i * 32 + lane; s += m2[(size_t)row * 128 + c] * w[c]; }
#pragma unroll
  for (int o = 16; o > 0; o >>= 1) s += __shfl_xor(s, o, 32);
  if (lane == 0) out[row] = 1.f / (1.f + __expf(-(s + b3[0])));
}

// ---------------------------------------------------------------------------
// Elementwise / graph kernels
// ---------------------------------------------------------------------------
__global__ void k_zero_f32(float* p, size_t n) {
  size_t i = (size_t)blockIdx.x * 256 + threadIdx.x;
  if (i < n) p[i] = 0.f;
}
__global__ void k_f32_to_bf16(const float* __restrict__ s, bhalf* __restrict__ d, size_t n) {
  size_t i = (size_t)blockIdx.x * 256 + threadIdx.x;
  if (i < n) d[i] = (bhalf)s[i];
}
// d[l][n][k] = s[l][k][n] (gcn_w used as x @ W -> store as [N][K] for B frags)
__global__ void k_transpose_bf16(const float* __restrict__ s, bhalf* __restrict__ d, int layers) {
  size_t n = (size_t)layers * 65536;
  size_t i = (size_t)blockIdx.x * 256 + threadIdx.x;
  if (i >= n) return;
  size_t l = i >> 16;
  int nn = (int)((i >> 8) & 255), kk = (int)(i & 255);
  d[i] = (bhalf)s[(l << 16) + (size_t)kk * 256 + nn];
}
__global__ void k_degrees(const int* __restrict__ src, const int* __restrict__ dst,
                          float* dout, float* din, int E) {
  int i = blockIdx.x * 256 + threadIdx.x;
  if (i < E) { atomicAdd(&dout[src[i]], 1.f); atomicAdd(&din[dst[i]], 1.f); }
}
__global__ void k_deg_inv(float* d, int n) {
  int i = blockIdx.x * 256 + threadIdx.x;
  if (i < n) d[i] = rsqrtf(fmaxf(d[i], 1.f));   // clip(deg,1)^-0.5
}
__global__ void k_scale_bf16(const float* __restrict__ gh, const float* __restrict__ dinv,
                             bhalf* __restrict__ o, size_t n) {
  size_t i = (size_t)blockIdx.x * 256 + threadIdx.x;
  if (i < n) o[i] = (bhalf)(gh[i] * dinv[i >> 8]);
}
__global__ void k_relu_scale(float* gh, const float* __restrict__ dinv, size_t n) {
  size_t i = (size_t)blockIdx.x * 256 + threadIdx.x;
  if (i < n) { float v = gh[i] * dinv[i >> 8]; gh[i] = fmaxf(v, 0.f); }
}
// agg[dst[e]] += xw[src[e]]  (64 threads/edge, float4 each -> f32 atomics)
__global__ void k_scatter(const float* __restrict__ xw, const int* __restrict__ src,
                          const int* __restrict__ dst, float* agg, int E) {
  size_t i = (size_t)blockIdx.x * 256 + threadIdx.x;
  size_t total = (size_t)E * 64;
  if (i >= total) return;
  int e = (int)(i >> 6);
  int q = ((int)i & 63) * 4;
  const float4 v = *(const float4*)(xw + (size_t)src[e] * 256 + q);
  float* o = agg + (size_t)dst[e] * 256 + q;
  atomicAdd(o + 0, v.x); atomicAdd(o + 1, v.y);
  atomicAdd(o + 2, v.z); atomicAdd(o + 3, v.w);
}
__global__ void k_gather(const float* __restrict__ src, const int* __restrict__ idx,
                         float* __restrict__ dst, bhalf* __restrict__ dstb, size_t n) {
  size_t i = (size_t)blockIdx.x * 256 + threadIdx.x;
  if (i >= n) return;
  int row = (int)(i >> 8), c = (int)(i & 255);
  float v = src[(size_t)idx[row] * 256 + c];
  dst[i] = v; dstb[i] = (bhalf)v;
}
__global__ void k_extract0(const float* __restrict__ h, bhalf* __restrict__ d, int Bn) {
  int i = blockIdx.x * 256 + threadIdx.x;
  if (i >= Bn * 256) return;
  int b = i >> 8, c = i & 255;
  d[i] = (bhalf)h[((size_t)b * 64) * 256 + c];   // h[:,0,:]
}

// ---------------------------------------------------------------------------
// Host orchestration
// ---------------------------------------------------------------------------
static inline dim3 GS(size_t n) { return dim3((unsigned)((n + 255) / 256)); }

extern "C" void kernel_launch(void* const* d_in, const int* in_sizes, int n_in,
                              void* d_out, int out_size, void* d_ws, size_t ws_size,
                              hipStream_t stream) {
  const float* feat0   = (const float*)d_in[0];
  const float* feat1   = (const float*)d_in[1];
  const float* fc_w0   = (const float*)d_in[2];
  const float* fc_b0   = (const float*)d_in[3];
  const float* fc_w1   = (const float*)d_in[4];
  const float* fc_b1   = (const float*)d_in[5];
  const float* gcn_w   = (const float*)d_in[6];
  const int*   e_src   = (const int*)d_in[7];
  const int*   e_dst   = (const int*)d_in[8];
  const int*   dg_seqs = (const int*)d_in[9];
  const int*   pt_seqs = (const int*)d_in[10];
  const float* gt_wl   = (const float*)d_in[13];
  const float* gt_wr   = (const float*)d_in[14];
  const float* gt_al   = (const float*)d_in[15];
  const float* gt_ar   = (const float*)d_in[16];
  const float* gt_wf   = (const float*)d_in[17];
  const float* lng     = (const float*)d_in[18];
  const float* lnb     = (const float*)d_in[19];
  const float* pd_w    = (const float*)d_in[20];
  const float* pd_b    = (const float*)d_in[21];
  const float* pp_w    = (const float*)d_in[22];
  const float* pp_b    = (const float*)d_in[23];
  const float* ml_w1   = (const float*)d_in[24];
  const float* ml_b1   = (const float*)d_in[25];
  const float* ml_w2   = (const float*)d_in[26];
  const float* ml_b2   = (const float*)d_in[27];
  const float* ml_w3   = (const float*)d_in[28];
  const float* ml_b3   = (const float*)d_in[29];
  float* out = (float*)d_out;

  const int E  = in_sizes[7];             // 640000
  const int Bq = in_sizes[9] / 64;        // 1024
  const int R  = Bq * 64;                 // 65536
  const int NN = in_sizes[12];            // 40000
  const int M0 = in_sizes[0] / 512;       // 20000

  char* ws = (char*)d_ws;
  size_t off = 0;
  auto alloc = [&](size_t bytes) { size_t o = off; off += (bytes + 255) & ~(size_t)255; return o; };

  bhalf* wb_fc0 = (bhalf*)(ws + alloc((size_t)256 * 512 * 2));
  bhalf* wb_fc1 = (bhalf*)(ws + alloc((size_t)256 * 512 * 2));
  bhalf* wb_gcn = (bhalf*)(ws + alloc((size_t)2 * 65536 * 2));
  bhalf* wb_wl  = (bhalf*)(ws + alloc((size_t)8 * 65536 * 2));
  bhalf* wb_wr  = (bhalf*)(ws + alloc((size_t)8 * 65536 * 2));
  bhalf* wb_wf  = (bhalf*)(ws + alloc((size_t)8 * 65536 * 2));
  bhalf* wb_pd  = (bhalf*)(ws + alloc((size_t)2 * 65536 * 2));
  bhalf* wb_pp  = (bhalf*)(ws + alloc((size_t)2 * 65536 * 2));
  bhalf* wb_m1  = (bhalf*)(ws + alloc((size_t)256 * 512 * 2));
  bhalf* wb_m2  = (bhalf*)(ws + alloc((size_t)128 * 256 * 2));
  float* dinv_o = (float*)(ws + alloc((size_t)NN * 4));
  float* dinv_i = (float*)(ws + alloc((size_t)NN * 4));
  float* gh     = (float*)(ws + alloc((size_t)NN * 256 * 4));
  char*  xw_raw = ws + alloc((size_t)NN * 256 * 4);       // xw; reused as h_bf later
  float* xw     = (float*)xw_raw;
  bhalf* h_bf   = (bhalf*)xw_raw;                         // 33.5MB < 41MB  (GCN done)
  char*  xs_raw = ws + alloc((size_t)M0 * 512 * 2);       // featbf / xs_bf (same size)
  bhalf* featbf = (bhalf*)xs_raw;
  bhalf* xs_bf  = (bhalf*)xs_raw;
  float* h      = (float*)(ws + alloc((size_t)R * 256 * 4));
  float* fl     = (float*)(ws + alloc((size_t)R * 256 * 4));  // also wf-output tmp
  float* fr     = (float*)(ws + alloc((size_t)R * 256 * 4));
  bhalf* hsa_bf = (bhalf*)(ws + alloc((size_t)R * 256 * 2));
  bhalf* embt_bf = (bhalf*)(ws + alloc((size_t)Bq * 256 * 2));
  bhalf* e1_bf   = (bhalf*)(ws + alloc((size_t)Bq * 256 * 2));
  bhalf* cat_bf  = (bhalf*)(ws + alloc((size_t)Bq * 512 * 2));
  bhalf* m1_bf   = (bhalf*)(ws + alloc((size_t)Bq * 256 * 2));
  float* m2f     = (float*)(ws + alloc((size_t)Bq * 128 * 4));
  (void)ws_size; (void)n_in; (void)out_size;

  // ---- weight prep (f32 -> bf16, [N][K] layouts) ----
  k_f32_to_bf16<<<GS(256 * 512), 256, 0, stream>>>(fc_w0, wb_fc0, 256 * 512);
  k_f32_to_bf16<<<GS(256 * 512), 256, 0, stream>>>(fc_w1, wb_fc1, 256 * 512);
  k_transpose_bf16<<<GS((size_t)2 * 65536), 256, 0, stream>>>(gcn_w, wb_gcn, 2);
  k_f32_to_bf16<<<GS((size_t)8 * 65536), 256, 0, stream>>>(gt_wl, wb_wl, (size_t)8 * 65536);
  k_f32_to_bf16<<<GS((size_t)8 * 65536), 256, 0, stream>>>(gt_wr, wb_wr, (size_t)8 * 65536);
  k_f32_to_bf16<<<GS((size_t)8 * 65536), 256, 0, stream>>>(gt_wf, wb_wf, (size_t)8 * 65536);
  k_f32_to_bf16<<<GS((size_t)2 * 65536), 256, 0, stream>>>(pd_w, wb_pd, (size_t)2 * 65536);
  k_f32_to_bf16<<<GS((size_t)2 * 65536), 256, 0, stream>>>(pp_w, wb_pp, (size_t)2 * 65536);
  k_f32_to_bf16<<<GS(256 * 512), 256, 0, stream>>>(ml_w1, wb_m1, 256 * 512);
  k_f32_to_bf16<<<GS(128 * 256), 256, 0, stream>>>(ml_w2, wb_m2, 128 * 256);

  // ---- degrees ----
  k_zero_f32<<<GS(NN), 256, 0, stream>>>(dinv_o, NN);
  k_zero_f32<<<GS(NN), 256, 0, stream>>>(dinv_i, NN);
  k_degrees<<<GS(E), 256, 0, stream>>>(e_src, e_dst, dinv_o, dinv_i, E);
  k_deg_inv<<<GS(NN), 256, 0, stream>>>(dinv_o, NN);
  k_deg_inv<<<GS(NN), 256, 0, stream>>>(dinv_i, NN);

  // ---- input FC: gh[0:M0] = feat0 @ fc_w0^T + b0; gh[M0:2M0] = feat1 @ fc_w1^T + b1
  k_f32_to_bf16<<<GS((size_t)M0 * 512), 256, 0, stream>>>(feat0, featbf, (size_t)M0 * 512);
  gemm_bf16<<<dim3(2, (M0 + 63) / 64), 256, 0, stream>>>(
      featbf, wb_fc0, fc_b0, gh, nullptr, 0, M0, 256, 512, 0);
  k_f32_to_bf16<<<GS((size_t)M0 * 512), 256, 0, stream>>>(feat1, featbf, (size_t)M0 * 512);
  gemm_bf16<<<dim3(2, (M0 + 63) / 64), 256, 0, stream>>>(
      featbf, wb_fc1, fc_b1, gh + (size_t)M0 * 256, nullptr, 0, M0, 256, 512, 0);

  // ---- 2x GCN: relu( D_in^-1/2 * Adj * (D_out^-1/2 * gh) @ W ) ----
  for (int l = 0; l < 2; ++l) {
    k_scale_bf16<<<GS((size_t)NN * 256), 256, 0, stream>>>(gh, dinv_o, xs_bf, (size_t)NN * 256);
    gemm_bf16<<<dim3(2, (NN + 63) / 64), 256, 0, stream>>>(
        xs_bf, wb_gcn + (size_t)l * 65536, nullptr, xw, nullptr, 0, NN, 256, 256, 0);
    k_zero_f32<<<GS((size_t)NN * 256), 256, 0, stream>>>(gh, (size_t)NN * 256);
    k_scatter<<<GS((size_t)E * 64), 256, 0, stream>>>(xw, e_src, e_dst, gh, E);
    k_relu_scale<<<GS((size_t)NN * 256), 256, 0, stream>>>(gh, dinv_i, (size_t)NN * 256);
  }

  // ---- graph transformer per node type ----
  for (int t = 0; t < 2; ++t) {
    const int* seqs = t ? pt_seqs : dg_seqs;
    k_gather<<<GS((size_t)R * 256), 256, 0, stream>>>(gh, seqs, h, h_bf, (size_t)R * 256);
    for (int l = 0; l < 4; ++l) {
      const int i = t * 4 + l;
      gemm_bf16<<<dim3(2, R / 64), 256, 0, stream>>>(
          h_bf, wb_wl + (size_t)i * 65536, nullptr, fl, nullptr, 0, R, 256, 256, 0);
      gemm_bf16<<<dim3(2, R / 64), 256, 0, stream>>>(
          h_bf, wb_wr + (size_t)i * 65536, nullptr, fr, nullptr, 0, R, 256, 256, 0);
      attn_kernel<<<Bq * 2, 64, 0, stream>>>(fl, fr, gt_al + (size_t)i * 128,
                                             gt_ar + (size_t)i * 128, hsa_bf);
      gemm_bf16<<<dim3(2, R / 64), 256, 0, stream>>>(
          hsa_bf, wb_wf + (size_t)i * 65536, nullptr, fl, nullptr, 0, R, 256, 256, 0);
      ln_residual<<<R / 8, 256, 0, stream>>>(h, fl, lng + (size_t)i * 256,
                                             lnb + (size_t)i * 256, h_bf, R);
    }
    k_extract0<<<GS((size_t)Bq * 256), 256, 0, stream>>>(h, embt_bf, Bq);
    const bhalf* pw = t ? wb_pp : wb_pd;
    const float* pb = t ? pp_b : pd_b;
    gemm_bf16<<<dim3(2, Bq / 64), 256, 0, stream>>>(
        embt_bf, pw, pb, nullptr, e1_bf, 256, Bq, 256, 256, 2 /*elu*/);
    gemm_bf16<<<dim3(2, Bq / 64), 256, 0, stream>>>(
        e1_bf, pw + 65536, pb + 256, nullptr, cat_bf + t * 256, 512, Bq, 256, 256, 0);
  }

  // ---- predict MLP ----
  gemm_bf16<<<dim3(2, Bq / 64), 256, 0, stream>>>(
      cat_bf, wb_m1, ml_b1, nullptr, m1_bf, 256, Bq, 256, 512, 1 /*relu*/);
  gemm_bf16<<<dim3(1, Bq / 64), 256, 0, stream>>>(
      m1_bf, wb_m2, ml_b2, m2f, nullptr, 0, Bq, 128, 256, 1 /*relu*/);
  head_kernel<<<(Bq + 7) / 8, 256, 0, stream>>>(m2f, ml_w3, ml_b3, out, Bq);
}
